// IGMC_283467842579
// MI455X (gfx1250) — matrix operand order repaired and checked by reading, not compiled, not run on hardware
//
#include <hip/hip_runtime.h>
#include <hip/hip_bf16.h>

typedef __attribute__((ext_vector_type(16))) __bf16 v16bf;
typedef __attribute__((ext_vector_type(8)))  float  v8f;

#define RREL   5
#define DD     64
#define KTOT   384          // (RREL+1)*DD
#define KTILES 12           // KTOT / 32

// Native CDNA5 v_tanh_f32 when available (TRANS32 op), else libm fallback.
__device__ __forceinline__ float fast_tanh(float x) {
#if __has_builtin(__builtin_amdgcn_tanhf)
    return __builtin_amdgcn_tanhf(x);
#elif __has_builtin(__builtin_amdgcn_tanh_f32)
    return __builtin_amdgcn_tanh_f32(x);
#else
    return tanhf(x);
#endif
}

// ---------------- zero int buffer ----------------
__global__ void k_zero_i32(int* __restrict__ p, int n) {
    int i = blockIdx.x * blockDim.x + threadIdx.x;
    if (i < n) p[i] = 0;
}

// ---------------- per-(dst, rel) edge counts ----------------
__global__ void k_count(const int* __restrict__ dst, const int* __restrict__ et,
                        int* __restrict__ cnt, int E) {
    int e = blockIdx.x * blockDim.x + threadIdx.x;
    if (e < E) atomicAdd(&cnt[(long)dst[e] * RREL + et[e]], 1);
}

// ---------------- build bf16 weight stack in WMMA B-fragment layout ----------------
// Wstack rows 0..319 : W_r[k][j] = comp[l,r,0]*basis[l,0,k,j] + comp[l,r,1]*basis[l,1,k,j]
// Wstack rows 320..383 : root[l][k][j]
// wfrag[((kt*4+ct)*32 + lane)*16 + e] = Wstack[kt*32 + (lane&16?8:0) + (e<8?e:e+8)][ct*16 + (lane&15)]
__global__ void k_build_w(const float* __restrict__ basis, const float* __restrict__ comp,
                          const float* __restrict__ root, __bf16* __restrict__ wfrag,
                          int l, int R, int NB) {
    int idx = blockIdx.x * blockDim.x + threadIdx.x;
    if (idx >= KTILES * 4 * 32 * 16) return;
    int e    =  idx        & 15;
    int lane = (idx >> 4)  & 31;
    int ct   = (idx >> 9)  & 3;
    int kt   =  idx >> 11;
    int k    = kt * 32 + ((lane & 16) ? 8 : 0) + (e < 8 ? e : e + 8);
    int col  = ct * 16 + (lane & 15);
    float v;
    if (k < RREL * DD) {
        int r = k / DD, kk = k % DD;
        const float* c  = comp + ((long)l * R + r) * NB;
        float b0 = basis[(((long)l * NB + 0) * DD + kk) * DD + col];
        float b1 = basis[(((long)l * NB + 1) * DD + kk) * DD + col];
        v = c[0] * b0 + c[1] * b1;
    } else {
        int kk = k - RREL * DD;
        v = root[((long)l * DD + kk) * DD + col];
    }
    wfrag[idx] = (__bf16)v;
}

// ---------------- stage xbuf = [ zeros(N,320) | h(N,64) ] ----------------
__global__ void k_prepare(const float* __restrict__ h, float* __restrict__ xbuf,
                          int Nreal, int Npad) {
    int idx = blockIdx.x * blockDim.x + threadIdx.x;
    int n = idx / 96, q = idx % 96;          // 96 float4 per 384-float row
    if (n >= Npad) return;
    float4 v = make_float4(0.f, 0.f, 0.f, 0.f);
    if (q >= 80 && n < Nreal)
        v = *(const float4*)(h + (long)n * DD + (q - 80) * 4);
    ((float4*)(xbuf + (long)n * KTOT))[q] = v;
}

// ---------------- scatter-add normalized messages into agg region of xbuf ----------------
__global__ void k_scatter(const float* __restrict__ h, const int* __restrict__ src,
                          const int* __restrict__ dst, const int* __restrict__ et,
                          const int* __restrict__ cnt, float* __restrict__ xbuf, int E) {
    long idx = (long)blockIdx.x * blockDim.x + threadIdx.x;
    long e = idx >> 4;                        // 16 threads/edge, 4 floats each
    if (e >= E) return;
    int part = (int)(idx & 15);
    int s = src[e], d = dst[e], r = et[e];
    int c = cnt[(long)d * RREL + r];
    float inv = 1.0f / (float)(c > 0 ? c : 1);
    float4 hv = *(const float4*)(h + (long)s * DD + part * 4);
    float* out = xbuf + (long)d * KTOT + r * DD + part * 4;
    atomicAdd(out + 0, hv.x * inv);
    atomicAdd(out + 1, hv.y * inv);
    atomicAdd(out + 2, hv.z * inv);
    atomicAdd(out + 3, hv.w * inv);
}

// ---------------- fused [Npad,384] @ [384,64] bf16-WMMA GEMM + bias + tanh ----------------
// One wave computes one 16-row x 64-col output tile: 4 col-tiles x 12 K-steps of
// v_wmma_f32_16x16x32_bf16 with f32 accumulation.
__global__ void k_gemm_tanh(const float* __restrict__ xbuf, const __bf16* __restrict__ wfrag,
                            const float* __restrict__ bias, float* __restrict__ hnext,
                            float* __restrict__ states, int tiles, int l, int L) {
    int wave = (blockIdx.x * blockDim.x + threadIdx.x) >> 5;
    int lane = threadIdx.x & 31;
    if (wave >= tiles) return;
    int m0 = wave * 16;
    int row = m0 + (lane & 15);
    int kbase = (lane & 16) ? 8 : 0;          // A-matrix lane-half K offset
    const float* xr = xbuf + (long)row * KTOT;

    v8f acc[4] = {};
    for (int kt = 0; kt < KTILES; ++kt) {
        int kb = kt * 32;
        // A fragment: 16x32 bf16; lane row = lane&15; elements 0..7 -> K=kbase..kbase+7,
        // elements 8..15 -> K=16+kbase..16+kbase+7 (per ISA 16-bit A layout).
        const float* p0 = xr + kb + kbase;
        const float* p1 = xr + kb + 16 + kbase;
        v16bf a;
#pragma unroll
        for (int e = 0; e < 8; ++e) {
            a[e]     = (__bf16)p0[e];
            a[8 + e] = (__bf16)p1[e];
        }
#pragma unroll
        for (int c = 0; c < 4; ++c) {
            v16bf b = *(const v16bf*)(wfrag + (((long)(kt * 4 + c) * 32 + lane) << 4));
            acc[c] = __builtin_amdgcn_wmma_f32_16x16x32_bf16(
                false, a, false, b, (short)0, acc[c], false, false);
        }
    }

    // Epilogue: C layout -> VGPR j holds M = m0 + j + (lane<16?0:8), N = c*16 + (lane&15)
    int roff = (lane & 16) ? 8 : 0;
    int Dcat = L * DD;
#pragma unroll
    for (int c = 0; c < 4; ++c) {
        int col = c * 16 + (lane & 15);
        float bv = bias[col];
#pragma unroll
        for (int j = 0; j < 8; ++j) {
            int rr = m0 + j + roff;
            float v = fast_tanh(acc[c][j] + bv);
            hnext[(long)rr * DD + col] = v;
            states[(long)rr * Dcat + l * DD + col] = v;
        }
    }
}

// ---------------- readout MLP: relu([cat_u|cat_m] @ w1 + b1) @ w2 + b2 ----------------
__global__ void k_mlp(const float* __restrict__ states, const float* __restrict__ w1,
                      const float* __restrict__ b1, const float* __restrict__ w2,
                      const float* __restrict__ b2, float* __restrict__ out, int B, int L) {
    int g = blockIdx.x, j = threadIdx.x;
    int H = blockDim.x;                      // 128 hidden units
    int Dcat = L * DD;                       // 256
    const float* u = states + (long)g * Dcat;
    const float* m = states + (long)(B + g) * Dcat;
    float acc = b1[j];
    for (int k = 0; k < Dcat; ++k) acc = fmaf(u[k], w1[(long)k * H + j], acc);
    for (int k = 0; k < Dcat; ++k) acc = fmaf(m[k], w1[(long)(Dcat + k) * H + j], acc);
    acc = fmaxf(acc, 0.f) * w2[j];
    __shared__ float red[256];
    red[j] = acc;
    __syncthreads();
    for (int s = H >> 1; s > 0; s >>= 1) {
        if (j < s) red[j] += red[j + s];
        __syncthreads();
    }
    if (j == 0) out[g] = red[0] + b2[0];
}

extern "C" void kernel_launch(void* const* d_in, const int* in_sizes, int n_in,
                              void* d_out, int out_size, void* d_ws, size_t ws_size,
                              hipStream_t stream) {
    const float* x     = (const float*)d_in[0];
    const float* basis = (const float*)d_in[1];
    const float* comp  = (const float*)d_in[2];
    const float* root  = (const float*)d_in[3];
    const float* bias  = (const float*)d_in[4];
    const float* w1    = (const float*)d_in[5];
    const float* b1    = (const float*)d_in[6];
    const float* w2    = (const float*)d_in[7];
    const float* b2    = (const float*)d_in[8];
    const int*   src   = (const int*)d_in[9];
    const int*   dst   = (const int*)d_in[10];
    const int*   et    = (const int*)d_in[11];

    int N  = in_sizes[0] / DD;
    int E  = in_sizes[9];
    int NB = 2;
    int L  = in_sizes[1] / (NB * DD * DD);
    int R  = in_sizes[2] / (L * NB);
    int B  = out_size;                        // one score per enclosing subgraph
    int Npad = (N + 15) & ~15;

    // carve workspace
    char* ws = (char*)d_ws;
    size_t off = 0;
    auto carve = [&](size_t bytes) -> void* {
        void* p = ws + off;
        off += (bytes + 255) & ~(size_t)255;
        return p;
    };
    int*    cnt    = (int*)   carve((size_t)N * R * sizeof(int));
    float*  xbuf   = (float*) carve((size_t)Npad * KTOT * sizeof(float));
    float*  hA     = (float*) carve((size_t)Npad * DD * sizeof(float));
    float*  hB     = (float*) carve((size_t)Npad * DD * sizeof(float));
    float*  states = (float*) carve((size_t)Npad * L * DD * sizeof(float));
    __bf16* wfrag  = (__bf16*)carve((size_t)KTILES * 4 * 32 * 16 * sizeof(__bf16));

    // edge counts per (dst, relation) — layer-invariant
    k_zero_i32<<<(N * R + 255) / 256, 256, 0, stream>>>(cnt, N * R);
    k_count<<<(E + 255) / 256, 256, 0, stream>>>(dst, et, cnt, E);

    const float* hcur = x;
    float* hnext = hA;
    for (int l = 0; l < L; ++l) {
        k_build_w<<<(KTILES * 4 * 32 * 16 + 255) / 256, 256, 0, stream>>>(
            basis, comp, root, wfrag, l, R, NB);

        int pthreads = Npad * 96;
        k_prepare<<<(pthreads + 255) / 256, 256, 0, stream>>>(hcur, xbuf, N, Npad);

        long sthreads = (long)E * 16;
        k_scatter<<<(int)((sthreads + 255) / 256), 256, 0, stream>>>(
            hcur, src, dst, et, cnt, xbuf, E);

        int tiles = Npad / 16;
        int blocks = (tiles + 7) / 8;         // 8 waves (tiles) per 256-thread block
        k_gemm_tanh<<<blocks, 256, 0, stream>>>(
            xbuf, wfrag, bias + (long)l * DD, hnext, states, tiles, l, L);

        hcur  = hnext;
        hnext = (hnext == hA) ? hB : hA;
    }

    k_mlp<<<B, 128, 0, stream>>>(states, w1, b1, w2, b2, (float*)d_out, B, L);
}
